// GCN_79061757984816
// MI455X (gfx1250) — matrix operand order repaired and checked by reading, not compiled, not run on hardware
//
#include <hip/hip_runtime.h>

// ---------------------------------------------------------------------------
// GCN forward for MI455X (gfx1250, wave32, WMMA + async-LDS staging).
// B=8, N=2048, F=64, HID=32.
// Roofline: dominated by streaming A (3 x 128 MB fp32 @ 23.3 TB/s). A is
// read once per layer with NT hint, converted to bf16 hi/lo in-register,
// and multiplied against LDS-staged bf16 hi/lo Z^T via
// v_wmma_f32_16x16x32_bf16 (3-term split => ~fp32 accuracy, fp32 accum).
// Z^T chunks are staged with global_load_async_to_lds_b128 (ASYNCcnt),
// double-buffered to overlap DMA with WMMA.
// ---------------------------------------------------------------------------

#define GB 8
#define GN 2048
#define GF 64
#define GHID 32
#define GTOL 1e-7f
#define KCHUNK 256
#define NCHUNK (GN / KCHUNK)

typedef __attribute__((ext_vector_type(16))) __bf16 v16bf;
typedef __attribute__((ext_vector_type(8)))  float  v8f;
typedef __attribute__((ext_vector_type(4)))  float  f32x4;

__device__ __forceinline__ unsigned short f2bf_rne(float f) {
  unsigned u = __float_as_uint(f);
  unsigned r = u + 0x7FFFu + ((u >> 16) & 1u);
  return (unsigned short)(r >> 16);
}
__device__ __forceinline__ float bf2f(unsigned short s) {
  return __uint_as_float(((unsigned)s) << 16);
}

__device__ __forceinline__ v8f wmma_bf16(v16bf a, v16bf b, v8f c) {
  // 8 args: (neg_a, A, neg_b, B, c_mod, C, reuse_a, reuse_b)
  return __builtin_amdgcn_wmma_f32_16x16x32_bf16(false, a, false, b,
                                                 (short)0, c, false, false);
}

// --------------------------------------------------------------------------
// Kernel 1: per-row degree with self-loop fixup: dm = rsqrt(rowsum + add),
// add = (|diag| < TOL). One block per (b,n) row. NT loads: A has no reuse.
// --------------------------------------------------------------------------
__global__ void k_degree(const float* __restrict__ a,
                         float* __restrict__ dm, float* __restrict__ addf) {
  __shared__ float s[256];
  const int bid = blockIdx.x;          // 0 .. B*N-1
  const int b = bid / GN, n = bid % GN;
  const int t = threadIdx.x;
  const size_t base = ((size_t)b * GN + n) * GN;
  float acc = 0.f;
  for (int m = t; m < GN; m += 256)
    acc += __builtin_nontemporal_load(&a[base + m]);
  s[t] = acc;
  __syncthreads();
  for (int off = 128; off > 0; off >>= 1) {
    if (t < off) s[t] += s[t + off];
    __syncthreads();
  }
  if (t == 0) {
    float diag = a[base + n];
    float add = (fabsf(diag) < GTOL) ? 1.f : 0.f;
    float d = s[0] + add;
    dm[(size_t)b * GN + n] = rsqrtf(d);
    addf[(size_t)b * GN + n] = add;
  }
}

// --------------------------------------------------------------------------
// Kernel 2: Z = dm[n] * (Hin @ W).  Writes fp32 row-major Zrow (self-loop
// term) and bf16 hi/lo transposed Z^T[b][k][m] (WMMA B-operand source).
// --------------------------------------------------------------------------
__global__ void k_feat(const float* __restrict__ Hin, const float* __restrict__ W,
                       const float* __restrict__ dm, int Fin,
                       float* __restrict__ zrow,
                       unsigned short* __restrict__ zhi,
                       unsigned short* __restrict__ zlo) {
  __shared__ float Wl[GF * GHID];
  const int t = threadIdx.x;
  const int nw = Fin * GHID;
  for (int i = t; i < nw; i += 256) Wl[i] = W[i];
  __syncthreads();

  const int idx = blockIdx.x * 256 + t;       // covers B*N*HID
  const int b = idx / (GN * GHID);
  const int r = idx % (GN * GHID);
  const int n = r / GHID;
  const int k = r % GHID;

  const float* hrow = Hin + ((size_t)b * GN + n) * Fin;
  float sum = 0.f;
  for (int f = 0; f < Fin; ++f) sum += hrow[f] * Wl[f * GHID + k];
  const float z = dm[(size_t)b * GN + n] * sum;

  zrow[((size_t)b * GN + n) * GHID + k] = z;
  unsigned short hb = f2bf_rne(z);
  unsigned short lb = f2bf_rne(z - bf2f(hb));
  const size_t tbase = ((size_t)b * GHID + k) * GN + n;
  zhi[tbase] = hb;
  zlo[tbase] = lb;
}

// --------------------------------------------------------------------------
// Kernel 3: WMMA GEMM with async-LDS double-buffered Z^T staging.
// Each wave computes a 16x32 output tile over K=2048; block = 128 threads
// (4 waves) -> 64 rows; grid (N/64, B) = 256 blocks.
// Per K-chunk (256): stage 32x256 hi + lo bf16 (32 KB) into LDS via
// global_load_async_to_lds_b128, then 8 k-steps x 6 WMMA from LDS operands.
// --------------------------------------------------------------------------
__global__ void __launch_bounds__(128)
k_gcn_gemm(const float* __restrict__ a,
           const float* __restrict__ zrow,
           const unsigned short* __restrict__ zhi,
           const unsigned short* __restrict__ zlo,
           const float* __restrict__ dm, const float* __restrict__ addf,
           const float* __restrict__ bias, float* __restrict__ hout) {
  // [buf][hi/lo][col][m_local]: 2*2*32*256*2B = 64 KB
  __shared__ __align__(64) unsigned short ldsZ[2][2][GHID][KCHUNK];

  const int b = blockIdx.y;
  const int wave = threadIdx.x >> 5;
  const int lane = threadIdx.x & 31;
  const int lane_lo = lane & 15;
  const int half = lane >> 4;              // 0 or 1
  const int rowbase = blockIdx.x * 64 + wave * 16;

  // ---- async staging of one 32-col x 256-m hi+lo chunk (16 B128/thread) ----
  auto stage = [&](int c, int buf) {
#pragma unroll
    for (int it = 0; it < 8; ++it) {
      const int i = (int)threadIdx.x + it * 128;   // 0..1023 transfer id
      const int col = i >> 5;                       // 32 x 16B blocks per col
      const int j = i & 31;
      const size_t goff = ((size_t)b * GHID + col) * GN + (size_t)c * KCHUNK + j * 8;
      const unsigned long long gh = (unsigned long long)(const void*)(zhi + goff);
      const unsigned long long gl = (unsigned long long)(const void*)(zlo + goff);
      const unsigned lh = (unsigned)(unsigned long long)(const void*)&ldsZ[buf][0][col][j * 8];
      const unsigned ll = (unsigned)(unsigned long long)(const void*)&ldsZ[buf][1][col][j * 8];
      asm volatile("global_load_async_to_lds_b128 %0, %1, off"
                   :: "v"(lh), "v"(gh) : "memory");
      asm volatile("global_load_async_to_lds_b128 %0, %1, off"
                   :: "v"(ll), "v"(gl) : "memory");
    }
  };

  // A operand: lane row = rowbase+lane_lo; contiguous K range
  // [base8, base8+16), matching the B operand order.
  const int base8 = half * 16;
  const float* arow = a + ((size_t)b * GN + rowbase + lane_lo) * GN;
  const int koff = half * 16;              // B operand m-offset within k-step

  v8f acc0 = {};
  v8f acc1 = {};

  stage(0, 0);
  for (int c = 0; c < NCHUNK; ++c) {
    asm volatile("s_wait_asynccnt 0x0" ::: "memory");  // chunk c resident
    __syncthreads();                                    // all waves done w/ c-1
    if (c + 1 < NCHUNK) stage(c + 1, (c + 1) & 1);
    const int buf = c & 1;

    for (int kk2 = 0; kk2 < KCHUNK; kk2 += 32) {
      const int kk = c * KCHUNK + kk2;
      __builtin_prefetch(arow + kk + 64, 0, 0);

      f32x4 p0 = __builtin_nontemporal_load((const f32x4*)(arow + kk + base8));
      f32x4 p1 = __builtin_nontemporal_load((const f32x4*)(arow + kk + base8 + 4));
      f32x4 p2 = __builtin_nontemporal_load((const f32x4*)(arow + kk + base8 + 8));
      f32x4 p3 = __builtin_nontemporal_load((const f32x4*)(arow + kk + base8 + 12));

      float fa[16];
#pragma unroll
      for (int i = 0; i < 4; ++i) {
        fa[i] = p0[i]; fa[4 + i] = p1[i]; fa[8 + i] = p2[i]; fa[12 + i] = p3[i];
      }
      v16bf aHi, aLo;
#pragma unroll
      for (int i = 0; i < 16; ++i) {
        unsigned short hb = f2bf_rne(fa[i]);
        unsigned short lb = f2bf_rne(fa[i] - bf2f(hb));
        aHi[i] = __builtin_bit_cast(__bf16, hb);
        aLo[i] = __builtin_bit_cast(__bf16, lb);
      }

      // B operands from LDS (ds_load_b128 pairs)
      v16bf b0h = *(const v16bf*)&ldsZ[buf][0][lane_lo][kk2 + koff];
      v16bf b0l = *(const v16bf*)&ldsZ[buf][1][lane_lo][kk2 + koff];
      v16bf b1h = *(const v16bf*)&ldsZ[buf][0][16 + lane_lo][kk2 + koff];
      v16bf b1l = *(const v16bf*)&ldsZ[buf][1][16 + lane_lo][kk2 + koff];

      acc0 = wmma_bf16(aHi, b0h, acc0);
      acc0 = wmma_bf16(aLo, b0h, acc0);
      acc0 = wmma_bf16(aHi, b0l, acc0);
      acc1 = wmma_bf16(aHi, b1h, acc1);
      acc1 = wmma_bf16(aLo, b1h, acc1);
      acc1 = wmma_bf16(aHi, b1l, acc1);
    }
  }

  // Epilogue. C/D layout: VGPR r, lane: col=lane&15, row = r + half*8.
  const int col0 = lane_lo;
  const int col1 = lane_lo + 16;
  const float bias0 = bias[col0];
  const float bias1 = bias[col1];
#pragma unroll
  for (int r = 0; r < 8; ++r) {
    const int row = rowbase + r + half * 8;
    const size_t rn = (size_t)b * GN + row;
    const float dmv = dm[rn];
    const float adv = addf[rn];
    const float* zr = zrow + rn * GHID;
    float v0 = acc0[r] + adv * zr[col0];
    float v1 = acc1[r] + adv * zr[col1];
    v0 = fmaxf(dmv * v0 + bias0, 0.f);
    v1 = fmaxf(dmv * v1 + bias1, 0.f);
    hout[rn * GHID + col0] = v0;
    hout[rn * GHID + col1] = v1;
  }
}

// --------------------------------------------------------------------------
// Kernel 4: partial max-pool over node chunks of 256.
// Grid (N/256, B), block 256 (8 rows x 32 feats), coalesced reads.
// --------------------------------------------------------------------------
__global__ void k_pool_partial(const float* __restrict__ H,
                               float* __restrict__ part) {
  __shared__ float s[256];
  const int b = blockIdx.y, chunk = blockIdx.x, t = threadIdx.x;
  const int k = t & 31, nl = t >> 5;
  float m = -3.4e38f;
  for (int it = 0; it < 32; ++it) {
    const int n = chunk * 256 + it * 8 + nl;
    m = fmaxf(m, H[((size_t)b * GN + n) * GHID + k]);
  }
  s[t] = m;
  __syncthreads();
  if (t < 32) {
    float mm = s[t];
#pragma unroll
    for (int j = 1; j < 8; ++j) mm = fmaxf(mm, s[t + 32 * j]);
    part[((size_t)b * (GN / 256) + chunk) * GHID + t] = mm;
  }
}

// --------------------------------------------------------------------------
// Kernel 5: finish max-pool + MLP head. Grid B, block 64.
// --------------------------------------------------------------------------
__global__ void k_mlp(const float* __restrict__ part,
                      const float* __restrict__ Wf1, const float* __restrict__ bf1,
                      const float* __restrict__ Wf2, const float* __restrict__ bf2,
                      float* __restrict__ out) {
  __shared__ float gs[GHID];
  __shared__ float red[64];
  const int b = blockIdx.x, t = threadIdx.x;
  if (t < GHID) {
    float mm = -3.4e38f;
    for (int c = 0; c < GN / 256; ++c)
      mm = fmaxf(mm, part[((size_t)b * (GN / 256) + c) * GHID + t]);
    gs[t] = mm;
  }
  __syncthreads();
  float h = bf1[t];
  for (int i = 0; i < GHID; ++i) h += gs[i] * Wf1[i * 64 + t];
  h = fmaxf(h, 0.f);
  red[t] = h * Wf2[t];
  __syncthreads();
  if (t == 0) {
    float s = 0.f;
    for (int i = 0; i < 64; ++i) s += red[i];
    out[b] = s + bf2[0];
  }
}

// --------------------------------------------------------------------------
extern "C" void kernel_launch(void* const* d_in, const int* in_sizes, int n_in,
                              void* d_out, int out_size, void* d_ws, size_t ws_size,
                              hipStream_t stream) {
  const float* x   = (const float*)d_in[0];
  const float* a   = (const float*)d_in[1];
  const float* W1  = (const float*)d_in[2];
  const float* b1  = (const float*)d_in[3];
  const float* W2  = (const float*)d_in[4];
  const float* b2  = (const float*)d_in[5];
  const float* W3  = (const float*)d_in[6];
  const float* b3  = (const float*)d_in[7];
  const float* Wf1 = (const float*)d_in[8];
  const float* bf1 = (const float*)d_in[9];
  const float* Wf2 = (const float*)d_in[10];
  const float* bf2 = (const float*)d_in[11];
  float* out = (float*)d_out;

  char* ws = (char*)d_ws;
  constexpr size_t SZ_BN   = (size_t)GB * GN * 4;                 // 64 KB
  constexpr size_t SZ_ZROW = (size_t)GB * GN * GHID * 4;          // 2 MB
  constexpr size_t SZ_ZT   = (size_t)GB * GHID * GN * 2;          // 1 MB
  float*          dm   = (float*)(ws);
  float*          addf = (float*)(ws + SZ_BN);
  float*          zrow = (float*)(ws + 2 * SZ_BN);
  unsigned short* zhi  = (unsigned short*)(ws + 2 * SZ_BN + SZ_ZROW);
  unsigned short* zlo  = (unsigned short*)(ws + 2 * SZ_BN + SZ_ZROW + SZ_ZT);
  float*          H1   = (float*)(ws + 2 * SZ_BN + SZ_ZROW + 2 * SZ_ZT);
  float*          H2   = (float*)(ws + 2 * SZ_BN + 2 * SZ_ZROW + 2 * SZ_ZT);
  float*          part = (float*)(ws + 2 * SZ_BN + 3 * SZ_ZROW + 2 * SZ_ZT);
  // total ~8.2 MB of d_ws used.

  const dim3 gGemm(GN / 64, GB);
  const dim3 gPool(GN / 256, GB);

  // normalization factors
  k_degree<<<GB * GN, 256, 0, stream>>>(a, dm, addf);

  // layer 1: F=64 -> 32
  k_feat<<<(GB * GN * GHID) / 256, 256, 0, stream>>>(x, W1, dm, GF, zrow, zhi, zlo);
  k_gcn_gemm<<<gGemm, 128, 0, stream>>>(a, zrow, zhi, zlo, dm, addf, b1, H1);
  // layer 2: 32 -> 32
  k_feat<<<(GB * GN * GHID) / 256, 256, 0, stream>>>(H1, W2, dm, GHID, zrow, zhi, zlo);
  k_gcn_gemm<<<gGemm, 128, 0, stream>>>(a, zrow, zhi, zlo, dm, addf, b2, H2);
  // layer 3: 32 -> 32
  k_feat<<<(GB * GN * GHID) / 256, 256, 0, stream>>>(H2, W3, dm, GHID, zrow, zhi, zlo);
  k_gcn_gemm<<<gGemm, 128, 0, stream>>>(a, zrow, zhi, zlo, dm, addf, b3, H1);

  // global max pool + MLP head
  k_pool_partial<<<gPool, 256, 0, stream>>>(H1, part);
  k_mlp<<<GB, 64, 0, stream>>>(part, Wf1, bf1, Wf2, bf2, out);
}
